// Gemma4MoEBlock_30288109371940
// MI455X (gfx1250) — matrix operand order repaired and checked by reading, not compiled, not run on hardware
//
#include <hip/hip_runtime.h>

// Problem constants (match reference)
#define T_TOK 2048
#define K_TOP 8
#define E_EXP 32
#define H_DIM 2816
#define I_DIM 704
#define CAP   640          // (T*K/E)*5/4
#define NCHUNK 64          // routing chunks
#define CHUNK  (T_TOK * K_TOP / NCHUNK)   // 256 assignments per chunk

#define LDS_STRIDE 80      // padded row stride (bytes): 16B-aligned, conflict-free

#if defined(__AMDGCN__) && defined(__has_builtin)
#  if __has_builtin(__builtin_amdgcn_global_load_async_to_lds_b128)
#    define HAVE_ASYNC 1
#  else
#    define HAVE_ASYNC 0
#  endif
#else
#  define HAVE_ASYNC 0
#endif

// WMMA fragment types (gfx1250, wave32)
typedef __attribute__((ext_vector_type(16))) __bf16 v16bf;
typedef __attribute__((ext_vector_type(8)))  float  v8f;
typedef __attribute__((ext_vector_type(4)))  int    v4i;

union BF16Frag {
    v16bf v;
    unsigned short u[16];
    unsigned int   w[8];
};

__device__ __forceinline__ unsigned short f2bf(float f) {
    return __builtin_bit_cast(unsigned short, (__bf16)f);   // hw cvt, RNE
}
__device__ __forceinline__ float bf2f(unsigned short s) {
    unsigned int u = ((unsigned int)s) << 16;
    return __builtin_bit_cast(float, u);
}

__device__ __forceinline__ v8f wmma_bf16(v16bf a, v16bf b, v8f c) {
    return __builtin_amdgcn_wmma_f32_16x16x32_bf16(false, a, false, b,
                                                   (short)0, c, false, false);
}

// A fragment (16x32 bf16): elements 0..7 at p (16B), 8..15 at p+8 ushorts (16B).
__device__ __forceinline__ v16bf load_a16(const unsigned short* p) {
    BF16Frag r;
    uint4 q0 = *reinterpret_cast<const uint4*>(p);
    uint4 q1 = *reinterpret_cast<const uint4*>(p + 8);
    r.w[0] = q0.x; r.w[1] = q0.y; r.w[2] = q0.z; r.w[3] = q0.w;
    r.w[4] = q1.x; r.w[5] = q1.y; r.w[6] = q1.z; r.w[7] = q1.w;
    return r.v;
}
// B fragment: 16 contiguous bf16 (32B).
__device__ __forceinline__ v16bf load_b16(const unsigned short* p) {
    BF16Frag r;
    uint4 q0 = *reinterpret_cast<const uint4*>(p);
    uint4 q1 = *reinterpret_cast<const uint4*>(p + 8);
    r.w[0] = q0.x; r.w[1] = q0.y; r.w[2] = q0.z; r.w[3] = q0.w;
    r.w[4] = q1.x; r.w[5] = q1.y; r.w[6] = q1.z; r.w[7] = q1.w;
    return r.v;
}
// B fragment from LDS (32 contiguous bytes) -> ds_load_b128 x2
__device__ __forceinline__ v16bf load_b16_lds(const unsigned char* p) {
    BF16Frag r;
    uint4 q0 = *reinterpret_cast<const uint4*>(p);
    uint4 q1 = *reinterpret_cast<const uint4*>(p + 16);
    r.w[0] = q0.x; r.w[1] = q0.y; r.w[2] = q0.z; r.w[3] = q0.w;
    r.w[4] = q1.x; r.w[5] = q1.y; r.w[6] = q1.z; r.w[7] = q1.w;
    return r.v;
}

#if HAVE_ASYNC
__device__ __forceinline__ void async_cp16(const unsigned short* g, unsigned char* l) {
    __builtin_amdgcn_global_load_async_to_lds_b128(
        (__attribute__((address_space(1))) v4i*)(g),
        (__attribute__((address_space(3))) v4i*)(l),
        0, 0);
}
#endif

__device__ __forceinline__ void wait_async0() {
#if defined(__AMDGCN__)
#  if defined(__has_builtin) && __has_builtin(__builtin_amdgcn_s_wait_asynccnt)
    __builtin_amdgcn_s_wait_asynccnt(0);
#  else
    asm volatile("s_wait_asynccnt 0x0" ::: "memory");
#  endif
#endif
}

__device__ __forceinline__ float gelu_tanh(float x) {
    const float c0 = 0.7978845608028654f;   // sqrt(2/pi)
    const float c1 = 0.044715f;
    float x3 = x * x * x;
    return 0.5f * x * (1.0f + tanhf(c0 * (x + c1 * x3)));
}

// ---------------------------------------------------------------------------
// fp32 -> bf16 bulk convert (8 elements / thread), memory-bound, packed cvt.
// ---------------------------------------------------------------------------
__global__ void __launch_bounds__(256)
cvt_bf16_kernel(const float* __restrict__ s, unsigned short* __restrict__ d, int n8) {
    int i = blockIdx.x * blockDim.x + threadIdx.x;
    if (i >= n8) return;
    const float4* p = reinterpret_cast<const float4*>(s) + (size_t)i * 2;
    float4 f0 = p[0], f1 = p[1];
    union { unsigned short u[8]; uint4 q; } r;
    r.u[0] = f2bf(f0.x); r.u[1] = f2bf(f0.y); r.u[2] = f2bf(f0.z); r.u[3] = f2bf(f0.w);
    r.u[4] = f2bf(f1.x); r.u[5] = f2bf(f1.y); r.u[6] = f2bf(f1.z); r.u[7] = f2bf(f1.w);
    reinterpret_cast<uint4*>(d)[i] = r.q;
}

// ---------------------------------------------------------------------------
// Routing, 3 passes (positions match reference's flat-order running count).
// ---------------------------------------------------------------------------
__global__ void route_hist_kernel(const int* __restrict__ topk_idx,
                                  int* __restrict__ hist) {
    int b = blockIdx.x, e = threadIdx.x;
    int cnt = 0;
    int i0 = b * CHUNK;
    for (int i = 0; i < CHUNK; ++i) cnt += (topk_idx[i0 + i] == e) ? 1 : 0;
    hist[b * E_EXP + e] = cnt;
}

__global__ void route_scan_kernel(const int* __restrict__ hist,
                                  int* __restrict__ base,
                                  int* __restrict__ counts) {
    int e = threadIdx.x;
    int run = 0;
    for (int b = 0; b < NCHUNK; ++b) {
        base[b * E_EXP + e] = run;
        run += hist[b * E_EXP + e];
    }
    counts[e] = run < CAP ? run : CAP;
}

__global__ void route_scatter_kernel(const int* __restrict__ topk_idx,
                                     const int* __restrict__ base,
                                     int* __restrict__ rowtok,
                                     int* __restrict__ pos) {
    int b = blockIdx.x, e = threadIdx.x;
    int cnt = base[b * E_EXP + e];
    int i0 = b * CHUNK;
    for (int i = 0; i < CHUNK; ++i) {
        int idx = i0 + i;
        if (topk_idx[idx] == e) {
            if (cnt < CAP) { rowtok[e * CAP + cnt] = idx; pos[idx] = cnt; }
            else           { pos[idx] = -1; }
            ++cnt;
        }
    }
}

// ---------------------------------------------------------------------------
// Phase 2: fused gate/up GEMM + gelu(g)*u, bf16 WMMA, f32 accumulate.
// grid.x = E*(CAP/128), grid.y = I/32, block = 128 (4 waves).
// Wave: 32 rows x 32 cols (gate+up) = 8 accumulators, 8 WMMA / K-step.
// B tile (gate 32 rows + up 32 rows, 32 K) staged in LDS via async DMA,
// double-buffered, ASYNCcnt + barrier synchronized. A stays in global
// (wave-private token-gathered rows).
// ---------------------------------------------------------------------------
__global__ void __launch_bounds__(128)
mlp1_kernel(const unsigned short* __restrict__ hsbf,
            const unsigned short* __restrict__ wgbf,
            const unsigned short* __restrict__ wubf,
            const int* __restrict__ counts,
            const int* __restrict__ rowtok,
            unsigned short* __restrict__ hbuf) {
    const int MT = CAP / 128;            // 5
    int e  = blockIdx.x / MT;
    int mt = blockIdx.x % MT;
    int cnt = counts[e];
    if (mt * 128 >= cnt) return;

    int wave = threadIdx.x >> 5;
    int lane = threadIdx.x & 31;
    int ln   = lane & 15;
    int hi   = lane >> 4;
    int kbA  = hi ? 16 : 0;

    int c0 = mt * 128 + wave * 32;
    int r0 = c0 + ln;        int rc0 = r0 < cnt ? r0 : cnt - 1;
    int r1 = c0 + 16 + ln;   int rc1 = r1 < cnt ? r1 : cnt - 1;
    int tok0 = rowtok[e * CAP + rc0] >> 3;   // flat = t*K + k, K==8
    int tok1 = rowtok[e * CAP + rc1] >> 3;

    const unsigned short* pa0 = hsbf + (size_t)tok0 * H_DIM + kbA;
    const unsigned short* pa1 = hsbf + (size_t)tok1 * H_DIM + kbA;

    int n0 = blockIdx.y * 32;
    const int NIT = H_DIM / 32;          // 88

    v8f z = {0.f,0.f,0.f,0.f,0.f,0.f,0.f,0.f};
    v8f G00 = z, G01 = z, G10 = z, G11 = z;
    v8f U00 = z, U01 = z, U10 = z, U11 = z;

#if HAVE_ASYNC
    // B tile in LDS: 64 rows (gate 0..31, up 32..63) x 64B payload, stride 80B.
    __shared__ unsigned char smem[2 * 64 * LDS_STRIDE];   // 10240 B
    const int BUF = 64 * LDS_STRIDE;                      // 5120

    int tid = threadIdx.x;
    int rs  = tid >> 2;                  // staging row 0..31
    int cs  = (tid & 3) * 8;             // ushort col offset 0,8,16,24
    const unsigned short* sg = wgbf + ((size_t)e * I_DIM + n0 + rs) * H_DIM + cs;
    const unsigned short* su = wubf + ((size_t)e * I_DIM + n0 + rs) * H_DIM + cs;
    unsigned lofg = (unsigned)rs * LDS_STRIDE + (unsigned)cs * 2;
    unsigned lofu = (unsigned)(32 + rs) * LDS_STRIDE + (unsigned)cs * 2;

    // prologue: stage buffer 0
    async_cp16(sg, smem + lofg); sg += 32;
    async_cp16(su, smem + lofu); su += 32;
    wait_async0();
    __syncthreads();

    unsigned lg0 = (unsigned)ln * LDS_STRIDE + (unsigned)hi * 32;
    unsigned lg1 = (unsigned)(16 + ln) * LDS_STRIDE + (unsigned)hi * 32;
    unsigned lu0 = (unsigned)(32 + ln) * LDS_STRIDE + (unsigned)hi * 32;
    unsigned lu1 = (unsigned)(48 + ln) * LDS_STRIDE + (unsigned)hi * 32;

    int cur = 0;
    for (int i = 0; i < NIT; ++i) {
        if (i + 1 < NIT) {
            unsigned char* nb = smem + (cur ^ 1) * BUF;
            async_cp16(sg, nb + lofg); sg += 32;
            async_cp16(su, nb + lofu); su += 32;
        }
        const unsigned char* bb = smem + cur * BUF;
        v16bf a0 = load_a16(pa0);  pa0 += 32;
        v16bf a1 = load_a16(pa1);  pa1 += 32;
        v16bf g0 = load_b16_lds(bb + lg0);
        v16bf g1 = load_b16_lds(bb + lg1);
        v16bf u0 = load_b16_lds(bb + lu0);
        v16bf u1 = load_b16_lds(bb + lu1);
        G00 = wmma_bf16(a0, g0, G00);
        G01 = wmma_bf16(a0, g1, G01);
        G10 = wmma_bf16(a1, g0, G10);
        G11 = wmma_bf16(a1, g1, G11);
        U00 = wmma_bf16(a0, u0, U00);
        U01 = wmma_bf16(a0, u1, U01);
        U10 = wmma_bf16(a1, u0, U10);
        U11 = wmma_bf16(a1, u1, U11);
        wait_async0();
        __syncthreads();
        cur ^= 1;
    }
#else
    int kbB = hi ? 16 : 0;
    const unsigned short* pg0 = wgbf + ((size_t)e * I_DIM + n0 + ln) * H_DIM + kbB;
    const unsigned short* pg1 = wgbf + ((size_t)e * I_DIM + n0 + 16 + ln) * H_DIM + kbB;
    const unsigned short* pu0 = wubf + ((size_t)e * I_DIM + n0 + ln) * H_DIM + kbB;
    const unsigned short* pu1 = wubf + ((size_t)e * I_DIM + n0 + 16 + ln) * H_DIM + kbB;
    for (int kk = 0; kk < H_DIM; kk += 32) {
        v16bf a0 = load_a16(pa0);  pa0 += 32;
        v16bf a1 = load_a16(pa1);  pa1 += 32;
        v16bf g0 = load_b16(pg0);  pg0 += 32;
        v16bf g1 = load_b16(pg1);  pg1 += 32;
        v16bf u0 = load_b16(pu0);  pu0 += 32;
        v16bf u1 = load_b16(pu1);  pu1 += 32;
        G00 = wmma_bf16(a0, g0, G00);
        G01 = wmma_bf16(a0, g1, G01);
        G10 = wmma_bf16(a1, g0, G10);
        G11 = wmma_bf16(a1, g1, G11);
        U00 = wmma_bf16(a0, u0, U00);
        U01 = wmma_bf16(a0, u1, U01);
        U10 = wmma_bf16(a1, u0, U10);
        U11 = wmma_bf16(a1, u1, U11);
    }
#endif

    // h = gelu(g) * u -> bf16.  D element v: row = base + v + 8*hi, col = n + ln.
    unsigned short* hb = hbuf + ((size_t)e * CAP) * I_DIM;
#pragma unroll
    for (int v = 0; v < 8; ++v) {
        int ra = c0 + v + 8 * hi;
        int rb = ra + 16;
        hb[(size_t)ra * I_DIM + n0 + ln]      = f2bf(gelu_tanh(G00[v]) * U00[v]);
        hb[(size_t)ra * I_DIM + n0 + 16 + ln] = f2bf(gelu_tanh(G01[v]) * U01[v]);
        hb[(size_t)rb * I_DIM + n0 + ln]      = f2bf(gelu_tanh(G10[v]) * U10[v]);
        hb[(size_t)rb * I_DIM + n0 + 16 + ln] = f2bf(gelu_tanh(G11[v]) * U11[v]);
    }
}

// ---------------------------------------------------------------------------
// Phase 3: down projection. o[e,c,n] = sum_i h[e,c,i] * w_down[e,n,i]
// grid.x = E*(CAP/128), grid.y = H/32, block = 128; wave: 32 rows x 32 cols.
// B tile (32 rows x 32 K) LDS-staged via async DMA, double-buffered.
// ---------------------------------------------------------------------------
__global__ void __launch_bounds__(128)
mlp2_kernel(const unsigned short* __restrict__ hbuf,
            const unsigned short* __restrict__ wdbf,
            const int* __restrict__ counts,
            unsigned short* __restrict__ obuf) {
    const int MT = CAP / 128;
    int e  = blockIdx.x / MT;
    int mt = blockIdx.x % MT;
    int cnt = counts[e];
    if (mt * 128 >= cnt) return;

    int wave = threadIdx.x >> 5;
    int lane = threadIdx.x & 31;
    int ln   = lane & 15;
    int hi   = lane >> 4;
    int kbA  = hi ? 16 : 0;

    int c0 = mt * 128 + wave * 32;
    const unsigned short* pa0 = hbuf + ((size_t)e * CAP + c0 + ln) * I_DIM + kbA;
    const unsigned short* pa1 = hbuf + ((size_t)e * CAP + c0 + 16 + ln) * I_DIM + kbA;

    int n0 = blockIdx.y * 32;
    const int NIT = I_DIM / 32;          // 22

    v8f z = {0.f,0.f,0.f,0.f,0.f,0.f,0.f,0.f};
    v8f D00 = z, D01 = z, D10 = z, D11 = z;

#if HAVE_ASYNC
    __shared__ unsigned char smem[2 * 32 * LDS_STRIDE];   // 5120 B
    const int BUF = 32 * LDS_STRIDE;                      // 2560

    int tid = threadIdx.x;
    int rs  = tid >> 2;
    int cs  = (tid & 3) * 8;
    const unsigned short* sb = wdbf + ((size_t)e * H_DIM + n0 + rs) * I_DIM + cs;
    unsigned lofb = (unsigned)rs * LDS_STRIDE + (unsigned)cs * 2;

    async_cp16(sb, smem + lofb); sb += 32;
    wait_async0();
    __syncthreads();

    unsigned lb0 = (unsigned)ln * LDS_STRIDE + (unsigned)hi * 32;
    unsigned lb1 = (unsigned)(16 + ln) * LDS_STRIDE + (unsigned)hi * 32;

    int cur = 0;
    for (int i = 0; i < NIT; ++i) {
        if (i + 1 < NIT) {
            async_cp16(sb, smem + (cur ^ 1) * BUF + lofb); sb += 32;
        }
        const unsigned char* bb = smem + cur * BUF;
        v16bf a0 = load_a16(pa0);  pa0 += 32;
        v16bf a1 = load_a16(pa1);  pa1 += 32;
        v16bf b0 = load_b16_lds(bb + lb0);
        v16bf b1 = load_b16_lds(bb + lb1);
        D00 = wmma_bf16(a0, b0, D00);
        D01 = wmma_bf16(a0, b1, D01);
        D10 = wmma_bf16(a1, b0, D10);
        D11 = wmma_bf16(a1, b1, D11);
        wait_async0();
        __syncthreads();
        cur ^= 1;
    }
#else
    int kbB = hi ? 16 : 0;
    const unsigned short* pb0 = wdbf + ((size_t)e * H_DIM + n0 + ln) * I_DIM + kbB;
    const unsigned short* pb1 = wdbf + ((size_t)e * H_DIM + n0 + 16 + ln) * I_DIM + kbB;
    for (int kk = 0; kk < I_DIM; kk += 32) {
        v16bf a0 = load_a16(pa0);  pa0 += 32;
        v16bf a1 = load_a16(pa1);  pa1 += 32;
        v16bf b0 = load_b16(pb0);  pb0 += 32;
        v16bf b1 = load_b16(pb1);  pb1 += 32;
        D00 = wmma_bf16(a0, b0, D00);
        D01 = wmma_bf16(a0, b1, D01);
        D10 = wmma_bf16(a1, b0, D10);
        D11 = wmma_bf16(a1, b1, D11);
    }
#endif

    unsigned short* ob = obuf + ((size_t)e * CAP) * H_DIM;
#pragma unroll
    for (int v = 0; v < 8; ++v) {
        int ra = c0 + v + 8 * hi;
        int rb = ra + 16;
        ob[(size_t)ra * H_DIM + n0 + ln]      = f2bf(D00[v]);
        ob[(size_t)ra * H_DIM + n0 + 16 + ln] = f2bf(D01[v]);
        ob[(size_t)rb * H_DIM + n0 + ln]      = f2bf(D10[v]);
        ob[(size_t)rb * H_DIM + n0 + 16 + ln] = f2bf(D11[v]);
    }
}

// ---------------------------------------------------------------------------
// Phase 4: deterministic combine. out[t,h] = sum_k w[t,k] * o[e_k, pos_k, h]
// ---------------------------------------------------------------------------
__global__ void combine_kernel(const int* __restrict__ topk_idx,
                               const float* __restrict__ topk_w,
                               const int* __restrict__ pos,
                               const unsigned short* __restrict__ obuf,
                               float* __restrict__ out) {
    int t = blockIdx.x;
    int e_[K_TOP]; float w_[K_TOP]; int p_[K_TOP];
#pragma unroll
    for (int k = 0; k < K_TOP; ++k) {
        e_[k] = topk_idx[t * K_TOP + k];
        w_[k] = topk_w[t * K_TOP + k];
        p_[k] = pos[t * K_TOP + k];
    }
    for (int hh = threadIdx.x; hh < H_DIM; hh += blockDim.x) {
        float acc = 0.f;
#pragma unroll
        for (int k = 0; k < K_TOP; ++k) {
            if (p_[k] >= 0) {
                acc += w_[k] * bf2f(obuf[((size_t)(e_[k] * CAP + p_[k])) * H_DIM + hh]);
            }
        }
        out[(size_t)t * H_DIM + hh] = acc;
    }
}

// ---------------------------------------------------------------------------
// Workspace layout (bytes):
//   counts @ 0            (128)
//   rowtok @ 256          (81,920)
//   pos    @ 82,176       (65,536)
//   hist   @ 147,712      (8,192)
//   base   @ 155,904      (8,192)
//   hsbf   @ 164,096      (11,534,336)
//   wgbf   @ 11,698,432   (126,877,696)
//   wubf   @ 138,576,128  (126,877,696)
//   wdbf   @ 265,453,824  (126,877,696)
//   hbuf   @ 392,331,520  (28,835,840)
//   obuf   @ 421,167,360  (115,343,360)   total ~= 536.5 MB
// ---------------------------------------------------------------------------
extern "C" void kernel_launch(void* const* d_in, const int* in_sizes, int n_in,
                              void* d_out, int out_size, void* d_ws, size_t ws_size,
                              hipStream_t stream) {
    const float* hs  = (const float*)d_in[0];
    const int*   tki = (const int*)d_in[1];
    const float* tkw = (const float*)d_in[2];
    const float* wg  = (const float*)d_in[3];
    const float* wu  = (const float*)d_in[4];
    const float* wd  = (const float*)d_in[5];
    float* out = (float*)d_out;

    char* ws = (char*)d_ws;
    int* counts = (int*)(ws + 0);
    int* rowtok = (int*)(ws + 256);
    int* pos    = (int*)(ws + 82176);
    int* hist   = (int*)(ws + 147712);
    int* base   = (int*)(ws + 155904);
    unsigned short* hsbf = (unsigned short*)(ws + 164096ull);
    unsigned short* wgbf = (unsigned short*)(ws + 11698432ull);
    unsigned short* wubf = (unsigned short*)(ws + 138576128ull);
    unsigned short* wdbf = (unsigned short*)(ws + 265453824ull);
    unsigned short* hbuf = (unsigned short*)(ws + 392331520ull);
    unsigned short* obuf = (unsigned short*)(ws + 421167360ull);

    // bf16 pre-conversion (memory-bound, hw packed cvt)
    {
        int nw8 = (E_EXP * I_DIM * H_DIM) / 8;            // 7,929,856
        int nh8 = (T_TOK * H_DIM) / 8;                    //   720,896
        int bw  = (nw8 + 255) / 256, bh = (nh8 + 255) / 256;
        cvt_bf16_kernel<<<bw, 256, 0, stream>>>(wg, wgbf, nw8);
        cvt_bf16_kernel<<<bw, 256, 0, stream>>>(wu, wubf, nw8);
        cvt_bf16_kernel<<<bw, 256, 0, stream>>>(wd, wdbf, nw8);
        cvt_bf16_kernel<<<bh, 256, 0, stream>>>(hs, hsbf, nh8);
    }

    // routing (3 passes)
    route_hist_kernel<<<NCHUNK, E_EXP, 0, stream>>>(tki, hist);
    route_scan_kernel<<<1, E_EXP, 0, stream>>>(hist, base, counts);
    route_scatter_kernel<<<NCHUNK, E_EXP, 0, stream>>>(tki, base, rowtok, pos);

    dim3 g1(E_EXP * (CAP / 128), I_DIM / 32);   // 160 x 22
    mlp1_kernel<<<g1, 128, 0, stream>>>(hsbf, wgbf, wubf, counts, rowtok, hbuf);

    dim3 g2(E_EXP * (CAP / 128), H_DIM / 32);   // 160 x 88
    mlp2_kernel<<<g2, 128, 0, stream>>>(hbuf, wdbf, counts, obuf);

    combine_kernel<<<T_TOK, 256, 0, stream>>>(tki, tkw, pos, obuf, out);
}
